// PCFGVMap_6597069767234
// MI455X (gfx1250) — matrix operand order repaired and checked by reading, not compiled, not run on hardware
//
#include <hip/hip_runtime.h>
#include <cstdint>
#include <cstddef>

#define NTS 30
#define TS 60
#define STATES 90
#define NN 48
#define MROWS 64                  // padded split-count rows (zeroed above l)
#define NEGV -1000000000.0f
#define CPAD 128                  // C (right-state) dim padded 90 -> 128 so k = (B<<7)|C
#define KPAD (STATES * CPAD)      // 11520 : flattened (B,C) contraction length
#define KSTEPS (KPAD / 32)        // 360 wmma k-steps of K=32
#define MT_TILES 2                // 32 rows cover NT=30
#define ASWZ_STRIDE 512           // halfs per (mt,kb) A-fragment: 32 lanes * 16 halfs

typedef __attribute__((ext_vector_type(16))) _Float16 v16h;
typedef __attribute__((ext_vector_type(8)))  float    v8f;

// ---------------------------------------------------------------------------
// Init: alpha = NEG everywhere except diagonal terminal scores; build exp(rule)
// pre-swizzled into this code's WMMA A-fragment layout (16-bit A 16x32, contiguous k order; B uses the same):
//   lane = {half[1], Mrow[4]}, element e:
//   K = kb*32 + half*16 + e
// so the GEMM kernel's A-load is one contiguous v16h per lane.
// ---------------------------------------------------------------------------
__global__ void pcfg_init(const float* __restrict__ unary,
                          const float* __restrict__ rule,
                          float* __restrict__ alpha,
                          _Float16* __restrict__ aswz) {
  int tid = blockIdx.x * blockDim.x + threadIdx.x;
  int nth = gridDim.x * blockDim.x;

  const int atot = NN * NN * STATES;
  for (int i = tid; i < atot; i += nth) {
    int st  = i % STATES;
    int pos = i / STATES;
    int c   = pos % NN;
    int r   = pos / NN;
    float v = NEGV;
    if (r == c && st >= NTS) v = unary[r * TS + (st - NTS)];
    alpha[i] = v;
  }

  const int swtot = MT_TILES * KSTEPS * ASWZ_STRIDE;   // 368640 halfs
  for (int i = tid; i < swtot; i += nth) {
    int e    = i & 15;
    int lane = (i >> 4) & 31;
    int kb   = (i >> 9) % KSTEPS;
    int mt   = i / (KSTEPS * ASWZ_STRIDE);
    int half = lane >> 4;
    int A    = mt * 16 + (lane & 15);
    int v    = e >> 1;
    int kin  = half * 16 + e; (void)v;
    int K    = kb * 32 + kin;             // < 11520
    int B    = K >> 7;                    // < 90
    int C    = K & (CPAD - 1);
    float val = 0.0f;
    if (A < NTS && C < STATES)
      val = __expf(rule[((size_t)A * STATES + B) * STATES + C]);
    aswz[i] = (_Float16)val;
  }
}

// ---------------------------------------------------------------------------
// One diagonal l: one block (128 thr = 4 waves) per span start s.
//   D[30 x l] = ExpRule[30 x 11520] x O^T, O_m[k] = expL_m[k>>7] * expR_m[k&127]
// B-fragments are built in registers: each lane needs 16 consecutive k at a
// 16-aligned base -> single B index -> one sL scalar * one v16h of sR.
// All 4 waves unconditionally run 2 output tiles (cols m = tile*16+lane%16,
// rows padded/zeroed to MROWS=64), so the k-loop is branch-free and EXEC is
// statically all-ones at every WMMA.
// ---------------------------------------------------------------------------
__global__ __launch_bounds__(128) void pcfg_span(const _Float16* __restrict__ aswz,
                                                 float* __restrict__ alpha,
                                                 int l) {
  __shared__ __align__(32) _Float16 sL[MROWS][96];    // exp(left - maxL), zero-padded
  __shared__ __align__(32) _Float16 sR[MROWS][CPAD];  // exp(right - maxR), zero-padded
  __shared__ float sScale[MROWS];
  __shared__ float sRes[32][NN];                      // inner_m[A]

  const int s    = blockIdx.x;
  const int tid  = threadIdx.x;
  const int lane = tid & 31;
  const int wave = tid >> 5;
  const int half = lane >> 4;
  const int col  = lane & 15;

  // Per-split max-shifted exponentials; rows m >= l zeroed so dead tiles
  // accumulate exact zeros.
  if (tid < MROWS) {
    const int m = tid;
    if (m < l) {
      const float* left  = alpha + ((size_t)s * NN + (s + m)) * STATES;
      const float* right = alpha + ((size_t)(s + m + 1) * NN + (s + l)) * STATES;
      float mL = NEGV, mR = NEGV;
      for (int i = 0; i < STATES; ++i) mL = fmaxf(mL, left[i]);
      for (int i = 0; i < STATES; ++i) mR = fmaxf(mR, right[i]);
      for (int i = 0; i < 96; ++i)
        sL[m][i] = (i < STATES) ? (_Float16)__expf(left[i] - mL) : (_Float16)0.0f;
      for (int i = 0; i < CPAD; ++i)
        sR[m][i] = (i < STATES) ? (_Float16)__expf(right[i] - mR) : (_Float16)0.0f;
      sScale[m] = mL + mR;
    } else {
      for (int i = 0; i < 96; ++i)  sL[m][i] = (_Float16)0.0f;
      for (int i = 0; i < CPAD; ++i) sR[m][i] = (_Float16)0.0f;
      sScale[m] = NEGV;
    }
  }
  __syncthreads();

  const int t0 = wave, t1 = wave + 4;       // this wave's two output tiles
  const int m0 = (t0 >> 1) * 16 + col;      // output column (split index), < 64
  const int m1 = (t1 >> 1) * 16 + col;
  const _Float16* a0base = aswz + (size_t)(t0 & 1) * KSTEPS * ASWZ_STRIDE + lane * 16;
  const _Float16* a1base = aswz + (size_t)(t1 & 1) * KSTEPS * ASWZ_STRIDE + lane * 16;

  v8f acc0 = {};
  v8f acc1 = {};

  for (int B = 0; B < STATES; ++B) {
    const _Float16 lv0 = sL[m0][B];
    const _Float16 lv1 = sL[m1][B];
#pragma unroll
    for (int cs = 0; cs < 4; ++cs) {
      const int kb = B * 4 + cs;            // global k-step, K = kb*32 + half*16 + e
      const int C  = cs * 32 + half * 16;   // 16-aligned, C+15 <= 127
      v16h a0 = *(const v16h*)(a0base + (size_t)kb * ASWZ_STRIDE);
      v16h b0 = (*(const v16h*)&sR[m0][C]) * lv0;
      acc0 = __builtin_amdgcn_wmma_f32_16x16x32_f16(false, a0, false, b0,
                                                    (short)0, acc0, false, false);
      v16h a1 = *(const v16h*)(a1base + (size_t)kb * ASWZ_STRIDE);
      v16h b1 = (*(const v16h*)&sR[m1][C]) * lv1;
      acc1 = __builtin_amdgcn_wmma_f32_16x16x32_f16(false, a1, false, b1,
                                                    (short)0, acc1, false, false);
    }
  }

  // Spill live accumulators: C/D layout => cols = lane&15, rows v + 8*half.
  // Divergence is fine here (all WMMAs are done).
  if (m0 < l) {
    int rbase = (t0 & 1) * 16 + (half ? 8 : 0);
    for (int v = 0; v < 8; ++v) sRes[rbase + v][m0] = acc0[v];
  }
  if (m1 < l) {
    int rbase = (t1 & 1) * 16 + (half ? 8 : 0);
    for (int v = 0; v < 8; ++v) sRes[rbase + v][m1] = acc1[v];
  }
  __syncthreads();

  // new[A] = M + log( sum_m inner_m[A] * exp(scale_m - M) )
  if (tid < NTS) {
    float M = NEGV;
    for (int m = 0; m < l; ++m) M = fmaxf(M, sScale[m]);
    float sum = 0.0f;
    for (int m = 0; m < l; ++m) sum += sRes[tid][m] * __expf(sScale[m] - M);
    alpha[((size_t)s * NN + (s + l)) * STATES + tid] = __logf(sum) + M;
  }
}

// ---------------------------------------------------------------------------
// log_Z = logsumexp(alpha[0, N-1, :NT] + root)
// ---------------------------------------------------------------------------
__global__ void pcfg_root(const float* __restrict__ alpha,
                          const float* __restrict__ root,
                          float* __restrict__ out) {
  if (blockIdx.x == 0 && threadIdx.x == 0) {
    const float* a = alpha + ((size_t)0 * NN + (NN - 1)) * STATES;
    float M = NEGV;
    for (int A = 0; A < NTS; ++A) M = fmaxf(M, a[A] + root[A]);
    float sum = 0.0f;
    for (int A = 0; A < NTS; ++A) sum += __expf(a[A] + root[A] - M);
    out[0] = __logf(sum) + M;
  }
}

extern "C" void kernel_launch(void* const* d_in, const int* in_sizes, int n_in,
                              void* d_out, int out_size, void* d_ws, size_t ws_size,
                              hipStream_t stream) {
  (void)in_sizes; (void)n_in; (void)out_size; (void)ws_size;
  const float* unary = (const float*)d_in[0];
  const float* rule  = (const float*)d_in[1];
  const float* root  = (const float*)d_in[2];
  float* out   = (float*)d_out;
  float* alpha = (float*)d_ws;                                         // 829,440 B
  const size_t alphaBytes = (size_t)NN * NN * STATES * sizeof(float);  // 32B-aligned
  _Float16* aswz = (_Float16*)((char*)d_ws + alphaBytes);              // 737,280 B

  pcfg_init<<<240, 256, 0, stream>>>(unary, rule, alpha, aswz);
  for (int l = 1; l < NN; ++l)
    pcfg_span<<<NN - l, 128, 0, stream>>>(aswz, alpha, l);
  pcfg_root<<<1, 32, 0, stream>>>(alpha, root, out);
}
